// GraphEncoder_44470091383015
// MI455X (gfx1250) — matrix operand order repaired and checked by reading, not compiled, not run on hardware
//
#include <hip/hip_runtime.h>

// ---------------------------------------------------------------------------
// GAT (3-layer) inference for MI455X / gfx1250, wave32 + WMMA.
// ---------------------------------------------------------------------------

#define N_NODES  40000
#define N_EDGES  320000
#define E_TOT    (N_EDGES + N_NODES)   // self loops appended
#define N_GRAPHS 400
#define NEG_SLOPE 0.2f

typedef __attribute__((ext_vector_type(16))) _Float16 v16h;
typedef __attribute__((ext_vector_type(8)))  _Float16 h8;
typedef __attribute__((ext_vector_type(8)))  float    v8f;

// ------------------------- small helpers -----------------------------------

__device__ __forceinline__ int ford_enc(float f) {
    int i = __float_as_int(f);
    return i >= 0 ? i : (i ^ 0x7fffffff);
}
__device__ __forceinline__ float ford_dec(int i) {
    return __int_as_float(i >= 0 ? i : (i ^ 0x7fffffff));
}

__global__ void fill_u32(unsigned* __restrict__ p, unsigned v, long n) {
    long i = (long)blockIdx.x * blockDim.x + threadIdx.x;
    if (i < n) p[i] = v;
}

// ------------------------- layer 0 GEMM (K = 6) -----------------------------

__global__ void gemm_k6(const float* __restrict__ X, const float* __restrict__ W,
                        float* __restrict__ Out) {
    int n = blockIdx.x;
    float xr[6];
#pragma unroll
    for (int k = 0; k < 6; ++k) xr[k] = X[n * 6 + k];
    for (int m = threadIdx.x; m < 512; m += blockDim.x) {
        float acc = 0.f;
#pragma unroll
        for (int k = 0; k < 6; ++k) acc += xr[k] * W[k * 512 + m];
        Out[(long)n * 512 + m] = acc;
    }
}

// ---------------- W fp32[K,M] -> f16 transposed [M,K] -----------------------

__global__ void w_transpose_half(const float* __restrict__ W, _Float16* __restrict__ Wt,
                                 int K, int M) {
    int idx = blockIdx.x * blockDim.x + threadIdx.x;
    if (idx >= K * M) return;
    int k = idx / M, m = idx - k * M;              // coalesced read of W
    Wt[(long)m * K + k] = (_Float16)W[idx];
}

// ------------------------- WMMA GEMM ----------------------------------------
// Out[N,M] = X[N,K] * W[K,M]; Wt is W transposed to [M,K] in f16.
// Block: 128 threads = 4 waves; block tile 64 rows x 128 cols.
// Wave tile: 32 rows x 64 cols = 2x4 accumulators of v_wmma_f32_16x16x32_f16.
// Requires: N % 64 == 0, K % 32 == 0, M % 128 == 0 (uniform EXEC for WMMA).

__global__ __launch_bounds__(128)
void gemm_wmma(const float* __restrict__ X, const _Float16* __restrict__ Wt,
               float* __restrict__ Out, int K, int M) {
    const int lane   = threadIdx.x & 31;
    const int wave   = threadIdx.x >> 5;
    const int hi     = lane >> 4;          // lane half: 0 (lanes 0-15) / 1 (16-31)
    const int lm     = lane & 15;
    const int wr     = wave >> 1;
    const int wc     = wave & 1;
    const int rowB   = blockIdx.x * 64 + wr * 32;
    const int colB   = blockIdx.y * 128 + wc * 64;

    v8f acc[2][4];
#pragma unroll
    for (int t = 0; t < 2; ++t)
#pragma unroll
        for (int c = 0; c < 4; ++c) acc[t][c] = (v8f){};

    for (int k0 = 0; k0 < K; k0 += 32) {
        // ---- A fragments (16x32 f16, ISA layout), converted f32->f16 ----
        v16h a[2];
#pragma unroll
        for (int t = 0; t < 2; ++t) {
            const float* xr = X + (long)(rowB + t * 16 + lm) * K + k0;
            const int kA = hi * 8;
#pragma unroll
            for (int i = 0; i < 8; ++i) a[t][i]     = (_Float16)xr[kA + i];
#pragma unroll
            for (int i = 0; i < 8; ++i) a[t][8 + i] = (_Float16)xr[16 + kA + i];
        }
        // ---- B fragments (32x16 f16) from transposed-half W ----
#pragma unroll
        for (int c = 0; c < 4; ++c) {
            const _Float16* wp = Wt + (long)(colB + c * 16 + lm) * K + k0 + hi * 8;
            h8 b0 = *(const h8*)(wp);
            h8 b1 = *(const h8*)(wp + 16);
            v16h b;
#pragma unroll
            for (int i = 0; i < 8; ++i) { b[i] = b0[i]; b[8 + i] = b1[i]; }
#pragma unroll
            for (int t = 0; t < 2; ++t)
                acc[t][c] = __builtin_amdgcn_wmma_f32_16x16x32_f16(
                    false, a[t], false, b, (short)0, acc[t][c], false, false);
        }
    }

    // ---- store C (fp32 16x16 layout: VGPR r -> row r + 8*hi, col = lane%16) ----
#pragma unroll
    for (int t = 0; t < 2; ++t)
#pragma unroll
        for (int c = 0; c < 4; ++c) {
            float* op = Out + (long)(rowB + t * 16 + hi * 8) * M + colB + c * 16 + lm;
#pragma unroll
            for (int r = 0; r < 8; ++r) op[(long)r * M] = acc[t][c][r];
        }
}

// -------------------- per-node attention dot products -----------------------
// ssrc[n,h] = sum_c h[n,h,c]*asrc[h,c]; sdst likewise. Block = 128 threads.

template <int H, int C>
__global__ void row_dots(const float* __restrict__ Hm, const float* __restrict__ asrc,
                         const float* __restrict__ adst, float* __restrict__ ssrc,
                         float* __restrict__ sdst) {
    const int n = blockIdx.x;
    const int t = threadIdx.x;
    const float* hp = Hm + (long)n * H * C;
    float ps[H], pd[H];
#pragma unroll
    for (int h = 0; h < H; ++h) { ps[h] = 0.f; pd[h] = 0.f; }
#pragma unroll
    for (int h = 0; h < H; ++h)
        for (int c = t; c < C; c += 128) {
            float v = hp[h * C + c];
            ps[h] += v * asrc[h * C + c];
            pd[h] += v * adst[h * C + c];
        }
    __shared__ float red[128];
#pragma unroll
    for (int h = 0; h < H; ++h) {
        red[t] = ps[h]; __syncthreads();
        for (int s = 64; s > 0; s >>= 1) { if (t < s) red[t] += red[t + s]; __syncthreads(); }
        if (t == 0) ssrc[n * H + h] = red[0];
        __syncthreads();
        red[t] = pd[h]; __syncthreads();
        for (int s = 64; s > 0; s >>= 1) { if (t < s) red[t] += red[t + s]; __syncthreads(); }
        if (t == 0) sdst[n * H + h] = red[0];
        __syncthreads();
    }
}

// -------------------- edge scores + segment max -----------------------------

template <int H>
__global__ void edge_scores(const int* __restrict__ src, const int* __restrict__ dst,
                            const float* __restrict__ ssrc, const float* __restrict__ sdst,
                            float* __restrict__ esc, int* __restrict__ mi) {
    int idx = blockIdx.x * blockDim.x + threadIdx.x;
    if (idx >= E_TOT * H) return;
    int e = idx / H, h = idx - e * H;
    int s = (e < N_EDGES) ? src[e] : (e - N_EDGES);
    int d = (e < N_EDGES) ? dst[e] : (e - N_EDGES);
    float v = ssrc[s * H + h] + sdst[d * H + h];
    v = (v > 0.f) ? v : NEG_SLOPE * v;
    esc[idx] = v;
    atomicMax(&mi[d * H + h], ford_enc(v));
}

// -------------------- exp + segment sum -------------------------------------

template <int H>
__global__ void edge_exp(const int* __restrict__ dst, float* __restrict__ esc,
                         const int* __restrict__ mi, float* __restrict__ denom) {
    int idx = blockIdx.x * blockDim.x + threadIdx.x;
    if (idx >= E_TOT * H) return;
    int e = idx / H, h = idx - e * H;
    int d = (e < N_EDGES) ? dst[e] : (e - N_EDGES);
    float w = __expf(esc[idx] - ford_dec(mi[d * H + h]));
    esc[idx] = w;
    atomicAdd(&denom[d * H + h], w);
}

// -------------------- weighted scatter aggregation --------------------------
// One block per edge: Agg[dst] += alpha * h[src]; coalesced reads, f32 atomics.

template <int H, int C>
__global__ void aggregate(const int* __restrict__ src, const int* __restrict__ dst,
                          const float* __restrict__ esc, const float* __restrict__ denom,
                          const float* __restrict__ Hm, float* __restrict__ Agg) {
    const int e = blockIdx.x;
    const int s = (e < N_EDGES) ? src[e] : (e - N_EDGES);
    const int d = (e < N_EDGES) ? dst[e] : (e - N_EDGES);
    float alpha[H];
#pragma unroll
    for (int h = 0; h < H; ++h) alpha[h] = esc[e * H + h] / denom[d * H + h];
    const float* hp = Hm + (long)s * H * C;
    float*       op = Agg + (long)d * H * C;
    for (int idx = threadIdx.x; idx < H * C; idx += blockDim.x) {
        int h = idx / C;
        atomicAdd(&op[idx], alpha[h] * hp[idx]);
    }
}

// -------------------- bias (+ optional relu) --------------------------------

__global__ void bias_relu(float* __restrict__ A, const float* __restrict__ b,
                          int hcMask, long total, int relu) {
    long i = (long)blockIdx.x * blockDim.x + threadIdx.x;
    if (i >= total) return;
    float v = A[i] + b[(int)(i & hcMask)];
    A[i] = relu ? fmaxf(v, 0.f) : v;
}

// -------------------- global mean pool --------------------------------------

__global__ void pool_sum(const float* __restrict__ A, const int* __restrict__ batch,
                         float* __restrict__ out, float* __restrict__ counts) {
    const int n = blockIdx.x;
    const int g = batch[n];
    for (int c = threadIdx.x; c < 256; c += blockDim.x)
        atomicAdd(&out[g * 256 + c], A[(long)n * 256 + c]);
    if (threadIdx.x == 0) atomicAdd(&counts[g], 1.0f);
}

__global__ void pool_div(float* __restrict__ out, const float* __restrict__ counts) {
    int g = blockIdx.x;
    out[g * 256 + threadIdx.x] /= fmaxf(counts[g], 1.0f);
}

// ---------------------------------------------------------------------------
// Host-side layer driver
// ---------------------------------------------------------------------------

template <int H, int C>
static void gat_layer(hipStream_t stream, const float* Hm, const float* asrc,
                      const float* adst, const float* bias, const int* src,
                      const int* dst, float* ssrc, float* sdst, int* mi,
                      float* denom, float* esc, float* Agg, int relu) {
    const long nh  = (long)N_NODES * H;
    const long eh  = (long)E_TOT * H;
    const long nhc = (long)N_NODES * H * C;

    row_dots<H, C><<<N_NODES, 128, 0, stream>>>(Hm, asrc, adst, ssrc, sdst);
    fill_u32<<<(int)((nh + 255) / 256), 256, 0, stream>>>((unsigned*)mi, 0x80000000u, nh);
    fill_u32<<<(int)((nh + 255) / 256), 256, 0, stream>>>((unsigned*)denom, 0u, nh);
    edge_scores<H><<<(int)((eh + 255) / 256), 256, 0, stream>>>(src, dst, ssrc, sdst, esc, mi);
    edge_exp<H><<<(int)((eh + 255) / 256), 256, 0, stream>>>(dst, esc, mi, denom);
    fill_u32<<<(int)((nhc + 255) / 256), 256, 0, stream>>>((unsigned*)Agg, 0u, nhc);
    aggregate<H, C><<<E_TOT, 256, 0, stream>>>(src, dst, esc, denom, Hm, Agg);
    bias_relu<<<(int)((nhc + 255) / 256), 256, 0, stream>>>(Agg, bias, H * C - 1, nhc, relu);
}

extern "C" void kernel_launch(void* const* d_in, const int* in_sizes, int n_in,
                              void* d_out, int out_size, void* d_ws, size_t ws_size,
                              hipStream_t stream) {
    (void)in_sizes; (void)n_in; (void)out_size; (void)ws_size;

    const float* x     = (const float*)d_in[0];
    const int*   ei    = (const int*)d_in[1];
    const int*   src   = ei;
    const int*   dst   = ei + N_EDGES;
    const int*   batch = (const int*)d_in[2];
    const float* W0 = (const float*)d_in[3];
    const float* as0 = (const float*)d_in[4];
    const float* ad0 = (const float*)d_in[5];
    const float* b0  = (const float*)d_in[6];
    const float* W1 = (const float*)d_in[7];
    const float* as1 = (const float*)d_in[8];
    const float* ad1 = (const float*)d_in[9];
    const float* b1  = (const float*)d_in[10];
    const float* W2 = (const float*)d_in[11];
    const float* as2 = (const float*)d_in[12];
    const float* ad2 = (const float*)d_in[13];
    const float* b2  = (const float*)d_in[14];
    float* out = (float*)d_out;

    // ---- workspace carve-up (256B aligned) ----
    size_t off = 0;
    auto wsAlloc = [&](size_t bytes) -> void* {
        void* p = (char*)d_ws + off;
        off += (bytes + 255) & ~(size_t)255;
        return p;
    };
    float*    hbuf   = (float*)wsAlloc((size_t)N_NODES * 512 * 4);
    float*    aggbuf = (float*)wsAlloc((size_t)N_NODES * 512 * 4);
    _Float16* wt     = (_Float16*)wsAlloc((size_t)512 * 512 * 2);
    float*    ssrc   = (float*)wsAlloc((size_t)N_NODES * 4 * 4);
    float*    sdst   = (float*)wsAlloc((size_t)N_NODES * 4 * 4);
    int*      mi     = (int*)wsAlloc((size_t)N_NODES * 4 * 4);
    float*    denom  = (float*)wsAlloc((size_t)N_NODES * 4 * 4);
    float*    esc    = (float*)wsAlloc((size_t)E_TOT * 4 * 4);
    float*    counts = (float*)wsAlloc((size_t)N_GRAPHS * 4);

    // ---------------- layer 0: [N,6] @ [6,512], H=4, C=128, relu ----------------
    gemm_k6<<<N_NODES, 256, 0, stream>>>(x, W0, hbuf);
    gat_layer<4, 128>(stream, hbuf, as0, ad0, b0, src, dst,
                      ssrc, sdst, mi, denom, esc, aggbuf, /*relu=*/1);

    // ---------------- layer 1: [N,512] @ [512,512] (WMMA), H=4, C=128, relu ----
    w_transpose_half<<<(512 * 512 + 255) / 256, 256, 0, stream>>>(W1, wt, 512, 512);
    gemm_wmma<<<dim3(N_NODES / 64, 512 / 128), 128, 0, stream>>>(aggbuf, wt, hbuf, 512, 512);
    gat_layer<4, 128>(stream, hbuf, as1, ad1, b1, src, dst,
                      ssrc, sdst, mi, denom, esc, aggbuf, /*relu=*/1);

    // ---------------- layer 2: [N,512] @ [512,256] (WMMA), H=1, C=256 ----------
    w_transpose_half<<<(512 * 256 + 255) / 256, 256, 0, stream>>>(W2, wt, 512, 256);
    gemm_wmma<<<dim3(N_NODES / 64, 256 / 128), 128, 0, stream>>>(aggbuf, wt, hbuf, 512, 256);
    gat_layer<1, 256>(stream, hbuf, as2, ad2, b2, src, dst,
                      ssrc, sdst, mi, denom, esc, aggbuf, /*relu=*/0);

    // ---------------- global mean pool ----------------
    fill_u32<<<(N_GRAPHS * 256 + 255) / 256, 256, 0, stream>>>((unsigned*)out, 0u,
                                                               (long)N_GRAPHS * 256);
    fill_u32<<<(N_GRAPHS + 255) / 256, 256, 0, stream>>>((unsigned*)counts, 0u, N_GRAPHS);
    pool_sum<<<N_NODES, 256, 0, stream>>>(aggbuf, batch, out, counts);
    pool_div<<<N_GRAPHS, 256, 0, stream>>>(out, counts);
}
